// MoE_12197707120945
// MI455X (gfx1250) — compile-verified, matrix operand order checked by reading, not run on hardware
//
#include <hip/hip_runtime.h>
#include <hip/hip_bf16.h>
#include <math.h>

#define EMB 768
#define HID 1152
#define NE  8
#define T_TOK 4096
#define LDX 792   // padded bf16 stride for the 32x768 X tile (kills DS bank conflicts)
#define LDH 40    // padded bf16 stride for the 32x32 H chunk

// W1 fragment grid: 24 K-chunks (768/32) x 72 N-tiles (1152/16)
#define W1_KC 24
#define W1_NT 72
// W2 fragment grid: 36 K-chunks (1152/32) x 48 N-tiles (768/16)
#define W2_KC 36
#define W2_NT 48
// one fragment = 32 lanes x 8 dwords = 256 dwords = 1KB
#define FRAG_DW 256

typedef __attribute__((ext_vector_type(16))) __bf16   v16bf;
typedef __attribute__((ext_vector_type(8)))  float    v8f;
typedef __attribute__((ext_vector_type(8)))  unsigned v8u;
typedef __attribute__((ext_vector_type(4)))  unsigned u32x4;
typedef __attribute__((ext_vector_type(8)))  int      i32x8;
typedef __attribute__((ext_vector_type(4)))  int      i32x4;

#if __has_builtin(__builtin_amdgcn_tensor_load_to_lds) && \
    __has_builtin(__builtin_amdgcn_s_wait_tensorcnt)
#define USE_TDM_GATHER 1
#else
#define USE_TDM_GATHER 0
#endif

// ---------------- bf16 pack helper -----------------------------------------

__device__ __forceinline__ unsigned pack_bf16(float lo, float hi) {
#if __has_builtin(__builtin_amdgcn_cvt_pk_bf16_f32)
  auto p = __builtin_amdgcn_cvt_pk_bf16_f32(lo, hi);
  return __builtin_bit_cast(unsigned, p);
#else
  unsigned short a = __builtin_bit_cast(unsigned short, (__bf16)lo);
  unsigned short b = __builtin_bit_cast(unsigned short, (__bf16)hi);
  return ((unsigned)b << 16) | (unsigned)a;
#endif
}

// ---------------- fragment helpers (CDNA5 WMMA bf16 layouts, ISA 7.12.2) ----

// A matrix 16x32 (MxK) bf16 from a row-major bf16 tile in LDS.
__device__ __forceinline__ v16bf load_a_frag(const __bf16* tile, int stride,
                                             int lane, int kbase) {
  const int m = lane & 15, half = lane >> 4;
  const unsigned* row = (const unsigned*)(tile + m * stride);
  v8u r;
#pragma unroll
  for (int v = 0; v < 4; ++v)
    r[v] = row[(kbase + half * 8 + 2 * v) >> 1];
#pragma unroll
  for (int v = 0; v < 4; ++v)
    r[4 + v] = row[(kbase + 16 + half * 8 + 2 * v) >> 1];
  return __builtin_bit_cast(v16bf, r);
}

// Pre-swizzled B fragment: lane's 8 dwords are contiguous (32B aligned).
__device__ __forceinline__ v16bf load_b_frag_pre(const unsigned* frag, int lane) {
  v8u r = *(const v8u*)(frag + lane * 8);
  return __builtin_bit_cast(v16bf, r);
}

__device__ __forceinline__ v8f wmma_bf16(v16bf a, v16bf b, v8f c) {
  return __builtin_amdgcn_wmma_f32_16x16x32_bf16(false, a, false, b,
                                                 (short)0, c, false, false);
}

// ---------------- kernel 0: zero output + expert counters ------------------

__global__ void moe_zero_kernel(float* __restrict__ out, int n,
                                int* __restrict__ counts) {
  const int i = blockIdx.x * 256 + threadIdx.x;
  if (i < n)  out[i] = 0.0f;
  if (i < NE) counts[i] = 0;
}

// ---------------- kernel 1: weight swizzle f32 -> bf16 WMMA-B fragments ----

__global__ void moe_swizzle_kernel(const float* __restrict__ w,
                                   unsigned* __restrict__ wf,
                                   int KC, int NT, int ldn) {
  const size_t i = (size_t)blockIdx.x * 256 + threadIdx.x;  // dword index
  const int v    = (int)(i & 7);
  const int lane = (int)((i >> 3) & 31);
  const size_t f = i >> 8;            // fragment index = (e*KC + kc)*NT + nt
  const int nt   = (int)(f % NT);
  const size_t ek = f / NT;
  const int kc   = (int)(ek % KC);
  const int e    = (int)(ek / KC);

  const int n    = nt * 16 + (lane & 15);
  const int half = lane >> 4;
  const int k    = kc * 32 + half * 8 + ((v < 4) ? 2 * v : 16 + 2 * (v - 4));

  const float* src = w + (size_t)e * (size_t)(KC * 32) * ldn;
  wf[i] = pack_bf16(src[(size_t)k * ldn + n], src[(size_t)(k + 1) * ldn + n]);
}

// ---------------- kernel 2: router (softmax + top-2, build expert lists) ---

__launch_bounds__(256, 1)
__global__ void moe_router_kernel(const float* __restrict__ x,
                                  const float* __restrict__ rw,
                                  const float* __restrict__ rb,
                                  int* __restrict__ counts,
                                  int* __restrict__ idx_list,
                                  float* __restrict__ gate_list) {
  __shared__ float srw[EMB * NE];  // 24 KB
  for (int i = threadIdx.x; i < EMB * NE; i += 256) srw[i] = rw[i];
  __syncthreads();

  const int t = blockIdx.x * 256 + threadIdx.x;
  float acc[NE];
#pragma unroll
  for (int e = 0; e < NE; ++e) acc[e] = rb[e];

  const float* xr = x + (size_t)t * EMB;
  for (int k = 0; k < EMB; ++k) {
    const float xv = xr[k];
#pragma unroll
    for (int e = 0; e < NE; ++e) acc[e] += xv * srw[k * NE + e];
  }

  float mx = acc[0];
#pragma unroll
  for (int e = 1; e < NE; ++e) mx = fmaxf(mx, acc[e]);
  float s = 0.0f;
#pragma unroll
  for (int e = 0; e < NE; ++e) { acc[e] = __expf(acc[e] - mx); s += acc[e]; }
  const float inv = 1.0f / s;

  // top-2 (lowest index wins ties, matching jax.lax.top_k)
  int i0 = 0; float v0 = acc[0];
#pragma unroll
  for (int e = 1; e < NE; ++e) if (acc[e] > v0) { v0 = acc[e]; i0 = e; }
  int i1 = -1; float v1 = -1.0f;
#pragma unroll
  for (int e = 0; e < NE; ++e)
    if (e != i0 && acc[e] > v1) { v1 = acc[e]; i1 = e; }

  int p = atomicAdd(&counts[i0], 1);
  idx_list[i0 * T_TOK + p] = t; gate_list[i0 * T_TOK + p] = v0 * inv;
  p = atomicAdd(&counts[i1], 1);
  idx_list[i1 * T_TOK + p] = t; gate_list[i1 * T_TOK + p] = v1 * inv;
}

// ---------------- kernel 3: fused expert MLP over compacted token tiles ----
// block = 256 threads (8 wave32), tile = 32 tokens of one expert.
// X rows are gathered by the Tensor Data Mover in gather mode (2 descriptors
// of 16 row-indices each), staged f32 in LDS, converted once to the bf16 A
// tile. Double-buffered pre-swizzled B fragments feed the WMMA pipe.

__launch_bounds__(256, 1)
__global__ void moe_mlp_kernel(const float* __restrict__ x,
                               const unsigned* __restrict__ w1f,
                               const float* __restrict__ b1,
                               const unsigned* __restrict__ w2f,
                               const float* __restrict__ b2,
                               const int* __restrict__ counts,
                               const int* __restrict__ idx_list,
                               const float* __restrict__ gate_list,
                               float* __restrict__ out) {
  const int e    = blockIdx.y;
  const int tile = blockIdx.x;
  const int cnt  = counts[e];
  if (tile * 32 >= cnt) return;  // block-uniform early exit

  __shared__ __bf16 lds_x[32 * LDX];     // 49.5 KB bf16 X tile
  __shared__ __bf16 lds_h[32 * LDH];     // 2.5 KB  H chunk (bf16)
  __shared__ float  lds_p[4][16][16];    // 4 KB    phase-1 K-split partials
  __shared__ int    s_tok[32];
  __shared__ float  s_gate[32];
#if USE_TDM_GATHER
  __shared__ float  lds_xf[32 * EMB];    // 96 KB TDM staging (f32 rows)
#endif

  const int tid  = threadIdx.x;
  const int lane = tid & 31;
  const int wave = tid >> 5;
  const int n16  = lane & 15;
  const int hsel = lane >> 4;

  if (tid < 32) {
    const int  g  = tile * 32 + tid;
    const bool ok = g < cnt;
    s_tok[tid]  = ok ? idx_list[e * T_TOK + g]  : 0;
    s_gate[tid] = ok ? gate_list[e * T_TOK + g] : 0.0f;  // pad rows contribute 0
  }
  __syncthreads();

#if USE_TDM_GATHER
  // ---- TDM gather: 2 descriptors x 16 indexed rows of 768 f32 ------------
  if (wave == 0) {
    // D# group1 (shared): mask=0, data_size=4B, tensor_dim0=768,
    // tensor_dim1=4096, tile_dim0=768, tile_dim1=16 idx, dim0_stride=768
    const i32x8 g1 = {0x00020000, 768 << 16, 4096 << 16, 768 << 16,
                      16, 768, 0, 0};
    const i32x8 z8 = {0, 0, 0, 0, 0, 0, 0, 0};
    const unsigned long long gaddr = (unsigned long long)(uintptr_t)x;
    const unsigned ga_lo = (unsigned)gaddr;
    const unsigned ga_hi = ((unsigned)(gaddr >> 32) & 0x01FFFFFFu) |
                           (2u << 30);  // D#.type = 2
#pragma unroll
    for (int half2 = 0; half2 < 2; ++half2) {
      const int* tk = s_tok + half2 * 16;
      const unsigned laddr =
          (unsigned)(uintptr_t)(lds_xf + half2 * 16 * EMB);
      // group0: count=1, gather_mode=1, 16-bit indices, lds/global address
      const u32x4 g0 = {0x80000001u, laddr, ga_lo, ga_hi};
      const i32x4 g2 = {tk[0] | (tk[1] << 16),  tk[2] | (tk[3] << 16),
                        tk[4] | (tk[5] << 16),  tk[6] | (tk[7] << 16)};
      const i32x4 g3 = {tk[8] | (tk[9] << 16),  tk[10] | (tk[11] << 16),
                        tk[12] | (tk[13] << 16), tk[14] | (tk[15] << 16)};
      __builtin_amdgcn_tensor_load_to_lds(g0, g1, g2, g3, z8, 0);
    }
    __builtin_amdgcn_s_wait_tensorcnt(0);
  }
  __syncthreads();
  // ---- convert staged f32 rows -> bf16 A tile ----------------------------
  {
    const int r  = tid >> 3;
    const int c0 = (tid & 7) * 96;
    const float4* src = (const float4*)(lds_xf + r * EMB);
#pragma unroll
    for (int j = 0; j < 24; ++j) {
      const int c = c0 + j * 4;
      const float4 v = src[c >> 2];
      unsigned* d = (unsigned*)(lds_x + r * LDX + c);
      d[0] = pack_bf16(v.x, v.y);
      d[1] = pack_bf16(v.z, v.w);
    }
  }
#else
  // ---- fallback: direct global gather f32 -> bf16 ------------------------
  {
    const int r  = tid >> 3;
    const int c0 = (tid & 7) * 96;
    const float4* src = (const float4*)(x + (size_t)s_tok[r] * EMB);
#pragma unroll
    for (int j = 0; j < 24; ++j) {
      const int c = c0 + j * 4;
      const float4 v = src[c >> 2];
      unsigned* d = (unsigned*)(lds_x + r * LDX + c);
      d[0] = pack_bf16(v.x, v.y);
      d[1] = pack_bf16(v.z, v.w);
    }
  }
#endif
  __syncthreads();

  const float* b1e = b1 + e * HID;
  const float* b2e = b2 + e * EMB;

  const int p_tid = wave & 3;   // which 16x16 tile of the H chunk
  const int p_mi  = p_tid >> 1; // token-row tile (0/1)
  const int p_ni  = p_tid & 1;  // hid-col tile (0/1)
  const int p_kh  = wave >> 2;  // K half of 768
  const int col0  = wave * 96;  // phase-2 output column slice

  v8f yacc[2][6];
#pragma unroll
  for (int mi = 0; mi < 2; ++mi)
#pragma unroll
    for (int nt = 0; nt < 6; ++nt)
      yacc[mi][nt] = v8f{0.f, 0.f, 0.f, 0.f, 0.f, 0.f, 0.f, 0.f};

  for (int ch = 0; ch < HID / 32; ++ch) {
    const int hid0 = ch * 32;

    // ---- phase 1: partial H tile, K in [p_kh*384, p_kh*384+384) -----------
    // double-buffered B fragments: next load issues under current WMMA
    v8f hacc = v8f{0.f, 0.f, 0.f, 0.f, 0.f, 0.f, 0.f, 0.f};
    const __bf16* atile = lds_x + p_mi * 16 * LDX;
    const unsigned* bfrag1 =
        w1f + (((size_t)(e * W1_KC + p_kh * 12) * W1_NT) + (2 * ch + p_ni))
              * FRAG_DW;
    v16bf bcur = load_b_frag_pre(bfrag1, lane);
#pragma unroll
    for (int ks = 0; ks < 12; ++ks) {
      v16bf bnxt = bcur;
      if (ks < 11)
        bnxt = load_b_frag_pre(bfrag1 + (size_t)(ks + 1) * W1_NT * FRAG_DW, lane);
      const v16bf a = load_a_frag(atile, LDX, lane, p_kh * 384 + ks * 32);
      hacc = wmma_bf16(a, bcur, hacc);
      bcur = bnxt;
    }

    // ---- reduce the two K halves, bias + exact GELU, bf16 H into LDS -----
    if (p_kh == 1) {
#pragma unroll
      for (int r = 0; r < 8; ++r)
        lds_p[p_tid][r + 8 * hsel][n16] = hacc[r];
    }
    __syncthreads();
    if (p_kh == 0) {
#pragma unroll
      for (int r = 0; r < 8; ++r) {
        const int m = r + 8 * hsel;
        const float v = hacc[r] + lds_p[p_tid][m][n16]
                      + b1e[hid0 + p_ni * 16 + n16];
        const float g = 0.5f * v * (1.0f + erff(v * 0.70710678118f));
        lds_h[(p_mi * 16 + m) * LDH + p_ni * 16 + n16] = (__bf16)g;
      }
    }
    __syncthreads();

    // ---- phase 2: Y[., col slice] += H_chunk @ W2[chunk, col slice] -------
    const v16bf a0 = load_a_frag(lds_h, LDH, lane, 0);
    const v16bf a1 = load_a_frag(lds_h + 16 * LDH, LDH, lane, 0);
    const unsigned* bfrag2 =
        w2f + ((size_t)(e * W2_KC + ch) * W2_NT + wave * 6) * FRAG_DW;
    v16bf b2cur = load_b_frag_pre(bfrag2, lane);
#pragma unroll
    for (int nt = 0; nt < 6; ++nt) {
      v16bf b2nxt = b2cur;
      if (nt < 5)
        b2nxt = load_b_frag_pre(bfrag2 + (size_t)(nt + 1) * FRAG_DW, lane);
      yacc[0][nt] = wmma_bf16(a0, b2cur, yacc[0][nt]);
      yacc[1][nt] = wmma_bf16(a1, b2cur, yacc[1][nt]);
      b2cur = b2nxt;
    }
    __syncthreads();  // protect lds_h / lds_p reuse next chunk
  }

  // ---- epilogue: +b2, scale by gate, scatter-add to out ------------------
#pragma unroll
  for (int mi = 0; mi < 2; ++mi) {
#pragma unroll
    for (int nt = 0; nt < 6; ++nt) {
#pragma unroll
      for (int r = 0; r < 8; ++r) {
        const int m = mi * 16 + r + 8 * hsel;
        const int c = col0 + nt * 16 + n16;
        const float val = (yacc[mi][nt][r] + b2e[c]) * s_gate[m];
        atomicAdd(out + (size_t)s_tok[m] * EMB + c, val);
      }
    }
  }
}

// ---------------- launcher --------------------------------------------------

extern "C" void kernel_launch(void* const* d_in, const int* in_sizes, int n_in,
                              void* d_out, int out_size, void* d_ws, size_t ws_size,
                              hipStream_t stream) {
  (void)in_sizes; (void)n_in; (void)out_size; (void)ws_size;
  const float* x  = (const float*)d_in[0];
  const float* rw = (const float*)d_in[1];
  const float* rb = (const float*)d_in[2];
  const float* w1 = (const float*)d_in[3];
  const float* b1 = (const float*)d_in[4];
  const float* w2 = (const float*)d_in[5];
  const float* b2 = (const float*)d_in[6];
  float* out = (float*)d_out;

  // workspace layout (requires ~29 MB scratch):
  //   [0, 256)            int counts[8]
  //   [256, +128K)        int idx_list[8*4096]
  //   [.., +128K)         float gate_list[8*4096]
  //   [512K, +13.5M)      w1 bf16 fragments
  //   [.., +13.5M)        w2 bf16 fragments
  char*     ws        = (char*)d_ws;
  int*      counts    = (int*)ws;
  int*      idx_list  = (int*)(ws + 256);
  float*    gate_list = (float*)(ws + 256 + NE * T_TOK * sizeof(int));
  unsigned* w1f       = (unsigned*)(ws + (size_t)512 * 1024);
  const size_t w1f_dw = (size_t)NE * W1_KC * W1_NT * FRAG_DW;   // 3,538,944
  unsigned* w2f       = w1f + w1f_dw;
  const size_t w2f_dw = (size_t)NE * W2_KC * W2_NT * FRAG_DW;   // 3,538,944

  const int n = T_TOK * EMB;
  moe_zero_kernel<<<(n + 255) / 256, 256, 0, stream>>>(out, n, counts);
  moe_swizzle_kernel<<<(unsigned)(w1f_dw / 256), 256, 0, stream>>>(w1, w1f,
                                                                   W1_KC, W1_NT, HID);
  moe_swizzle_kernel<<<(unsigned)(w2f_dw / 256), 256, 0, stream>>>(w2, w2f,
                                                                   W2_KC, W2_NT, EMB);
  moe_router_kernel<<<T_TOK / 256, 256, 0, stream>>>(x, rw, rb, counts,
                                                     idx_list, gate_list);
  dim3 grid(T_TOK / 32, NE);
  moe_mlp_kernel<<<grid, 256, 0, stream>>>(x, w1f, b1, w2f, b2, counts,
                                           idx_list, gate_list, out);
}
